// RTDETRDecoder_37452114821802
// MI455X (gfx1250) — compile-verified, matrix operand order checked by reading, not run on hardware
//
#include <hip/hip_runtime.h>
#include <hip/hip_bf16.h>

#define CC 256
#define HH 8
#define HD 32
#define LL 6
#define FF 1024
#define NLV 4
#define PP 4
#define NCLS 80
#define BB 8
#define QQ 300
#define SS 8500   // 80*80+40*40+20*20+10*10
#define BQ (BB*QQ)          // 2400
#define MVAL (BB*SS)        // 68000

typedef __attribute__((ext_vector_type(16))) _Float16 v16h;
typedef __attribute__((ext_vector_type(8)))  _Float16 v8h;
typedef __attribute__((ext_vector_type(8)))  float    v8f;

// ---------------------------------------------------------------------------
// f32 -> f16 elementwise convert (activations)
// ---------------------------------------------------------------------------
__global__ void cvt_f16_kernel(const float* __restrict__ x, _Float16* __restrict__ o, int n) {
  int i = blockIdx.x * blockDim.x + threadIdx.x;
  if (i < n) o[i] = (_Float16)x[i];
}

// Pack weights: W[K,N] f32 -> Wt[Npad,K] f16 (transposed, N zero-padded to x16)
__global__ void pack_w_kernel(const float* __restrict__ W, _Float16* __restrict__ Wt,
                              int K, int N, int Npad) {
  int i = blockIdx.x * blockDim.x + threadIdx.x;
  if (i >= Npad * K) return;
  int k = i % K;
  int col = i / K;
  Wt[i] = (col < N) ? (_Float16)W[(size_t)k * N + col] : (_Float16)0.f;
}

// ---------------------------------------------------------------------------
// WMMA GEMM: Out[M,N] = act(A[M,K] @ W[K,N] + bias[N])
// A: f16 [M,K] row-major.  Wt: f16 [Npad,K] (pre-transposed, padded).
// One wave -> NT consecutive 16x16 tiles (NT is compile-time: no divergence,
// no exec-mask juggling, A fragment reused NT times per K-step).
// Launch guarantees tilesN % NT == 0. K mult of 32, M mult of 16.
// ---------------------------------------------------------------------------
template<int NT>
__global__ void wmma_gemm_kernel(const _Float16* __restrict__ A,
                                 const _Float16* __restrict__ Wt,
                                 const float* __restrict__ bias,
                                 float* __restrict__ Out,
                                 int M, int N, int K, int act) {
  const int wave = threadIdx.x >> 5;
  const int lane = threadIdx.x & 31;
  const int tilesN  = (N + 15) >> 4;
  const int nGroups = tilesN / NT;                  // exact by construction
  const int grp = blockIdx.x * 4 + wave;
  if (grp >= nGroups) return;                       // wave-uniform, before any WMMA
  const int tnBase = grp * NT;
  const int tm = blockIdx.y;
  const int half = lane >> 4;                       // 0 or 1
  const int lx   = lane & 15;
  const int row  = tm * 16 + lx;                    // A row (< M always)

  const _Float16* Arow = A + (size_t)row * K + (size_t)half * 8;
  const _Float16* Brow[NT];
#pragma unroll
  for (int t = 0; t < NT; ++t)
    Brow[t] = Wt + (size_t)((tnBase + t) * 16 + lx) * K + (size_t)half * 8;

  v8f acc[NT];
#pragma unroll
  for (int t = 0; t < NT; ++t) {
    const int col = (tnBase + t) * 16 + lx;
    float bval = 0.f;
    if (bias != nullptr && col < N) bval = bias[col];
#pragma unroll
    for (int r = 0; r < 8; ++r) acc[t][r] = bval;
  }

  for (int k0 = 0; k0 < K; k0 += 32) {
    // a[j] = A[row][k0 + 8*half + j], a[8+j] = A[row][k0 + 16 + 8*half + j]
    v8h alo = *(const v8h*)(Arow + k0);
    v8h ahi = *(const v8h*)(Arow + k0 + 16);
    v16h a;
#pragma unroll
    for (int j = 0; j < 8; ++j) { a[j] = alo[j]; a[8 + j] = ahi[j]; }
#pragma unroll
    for (int t = 0; t < NT; ++t) {
      v8h blo = *(const v8h*)(Brow[t] + k0);
      v8h bhi = *(const v8h*)(Brow[t] + k0 + 16);
      v16h b;
#pragma unroll
      for (int j = 0; j < 8; ++j) { b[j] = blo[j]; b[8 + j] = bhi[j]; }
      acc[t] = __builtin_amdgcn_wmma_f32_16x16x32_f16(false, a, false, b,
                                                      (short)0, acc[t], false, false);
    }
  }

#pragma unroll
  for (int t = 0; t < NT; ++t) {
    const int col = (tnBase + t) * 16 + lx;
#pragma unroll
    for (int r = 0; r < 8; ++r) {
      const int orow = tm * 16 + r + 8 * half;      // C/D layout: VGPR r -> M = r + 8*half
      if (col < N) {                                // only divergent when N==4
        float v = acc[t][r];
        if (act == 1) v = fmaxf(v, 0.f);
        Out[(size_t)orow * N + col] = v;
      }
    }
  }
}

// ---------------------------------------------------------------------------
// Elementwise helpers
// ---------------------------------------------------------------------------
__global__ void sigmoid_kernel(const float* __restrict__ x, float* __restrict__ o, int n) {
  int i = blockIdx.x * blockDim.x + threadIdx.x;
  if (i < n) o[i] = 1.f / (1.f + expf(-x[i]));
}

__global__ void add_kernel(const float* __restrict__ a, const float* __restrict__ b,
                           float* __restrict__ o, int n) {
  int i = blockIdx.x * blockDim.x + threadIdx.x;
  if (i < n) o[i] = a[i] + b[i];
}

// sine positional embedding of qsrc = ref * vr(level 0) -> pe[BQ, 2*C]
__global__ void pe_kernel(const float* __restrict__ ref, const float* __restrict__ vr,
                          float* __restrict__ pe) {
  int i = blockIdx.x * blockDim.x + threadIdx.x;
  const int n = BQ * 2 * CC;
  if (i >= n) return;
  int f  = i & 511;
  int bq = i >> 9;
  int b  = bq / QQ;
  int j  = f >> 7;
  int ff = f & 127;
  int p  = ff >> 1;
  float t   = powf(10000.f, (float)p * (1.f / 64.f));
  float pos = ref[bq * 4 + j] * vr[(b * NLV + 0) * 2 + (j & 1)];
  float x   = pos * 6.283185307179586f / t;
  pe[i] = (ff & 1) ? cosf(x) : sinf(x);
}

// fused residual layernorm over C=256: out = LN(x (+res)) * g + b
__global__ void ln_kernel(const float* __restrict__ x, const float* __restrict__ res,
                          const float* __restrict__ g, const float* __restrict__ be,
                          float* __restrict__ out) {
  const int row = blockIdx.x;
  const int c   = threadIdx.x;
  __shared__ float red[CC];
  float v = x[(size_t)row * CC + c];
  if (res != nullptr) v += res[(size_t)row * CC + c];
  red[c] = v; __syncthreads();
  for (int s = CC / 2; s > 0; s >>= 1) { if (c < s) red[c] += red[c + s]; __syncthreads(); }
  float mean = red[0] * (1.f / CC); __syncthreads();
  float d = v - mean;
  red[c] = d * d; __syncthreads();
  for (int s = CC / 2; s > 0; s >>= 1) { if (c < s) red[c] += red[c + s]; __syncthreads(); }
  float var = red[0] * (1.f / CC);
  out[(size_t)row * CC + c] = d * rsqrtf(var + 1e-5f) * g[c] + be[c];
}

// ---------------------------------------------------------------------------
// MHA softmax-attention core: qkv[BQ, 3C] -> ctx[BQ, C]. One block per (b,h,q).
// ---------------------------------------------------------------------------
__global__ void mha_attn_kernel(const float* __restrict__ qkv, float* __restrict__ ctx) {
  const int blk = blockIdx.x;
  const int qi = blk % QQ;
  const int bh = blk / QQ;
  const int h  = bh % HH;
  const int b  = bh / HH;
  const int tid = threadIdx.x;
  __shared__ float qs[HD];
  __shared__ float probs[QQ];
  __shared__ float red[128];
  if (tid < HD) qs[tid] = qkv[(size_t)(b * QQ + qi) * (3 * CC) + h * HD + tid];
  __syncthreads();
  const float scale = 0.17677669529663687f;   // 1/sqrt(32)
  float lmax = -1e30f;
  for (int k = tid; k < QQ; k += 128) {
    const float* kv = qkv + (size_t)(b * QQ + k) * (3 * CC) + CC + h * HD;
    float s = 0.f;
#pragma unroll
    for (int d = 0; d < HD; ++d) s += qs[d] * kv[d];
    s *= scale;
    probs[k] = s;
    lmax = fmaxf(lmax, s);
  }
  red[tid] = lmax; __syncthreads();
  for (int s = 64; s > 0; s >>= 1) { if (tid < s) red[tid] = fmaxf(red[tid], red[tid + s]); __syncthreads(); }
  const float m = red[0]; __syncthreads();
  float lsum = 0.f;
  for (int k = tid; k < QQ; k += 128) { float e = expf(probs[k] - m); probs[k] = e; lsum += e; }
  red[tid] = lsum; __syncthreads();
  for (int s = 64; s > 0; s >>= 1) { if (tid < s) red[tid] += red[tid + s]; __syncthreads(); }
  const float inv = 1.f / red[0];
  __syncthreads();
  if (tid < HD) {
    float acc = 0.f;
    for (int k = 0; k < QQ; ++k)
      acc += probs[k] * qkv[(size_t)(b * QQ + k) * (3 * CC) + 2 * CC + h * HD + tid];
    ctx[(size_t)(b * QQ + qi) * CC + h * HD + tid] = acc * inv;
  }
}

// softmax over 16 (NL*P) for deformable attention weights: [BQ*H, 16]
__global__ void softmax16_kernel(const float* __restrict__ in, float* __restrict__ out) {
  int i = blockIdx.x * blockDim.x + threadIdx.x;
  if (i >= BQ * HH) return;
  const float* p = in + (size_t)i * 16;
  float m = -1e30f;
#pragma unroll
  for (int j = 0; j < 16; ++j) m = fmaxf(m, p[j]);
  float e[16]; float s = 0.f;
#pragma unroll
  for (int j = 0; j < 16; ++j) { e[j] = expf(p[j] - m); s += e[j]; }
  const float inv = 1.f / s;
  float* o = out + (size_t)i * 16;
#pragma unroll
  for (int j = 0; j < 16; ++j) o[j] = e[j] * inv;
}

// ---------------------------------------------------------------------------
// Multi-scale deformable attention sampling. One block per (b,q,h), 32 threads (d).
// ---------------------------------------------------------------------------
__global__ void msda_kernel(const float* __restrict__ val,   // [B*S, C] as (b,s,h,d)
                            const float* __restrict__ off,   // [BQ, 256] = (h,l,p,2)
                            const float* __restrict__ aw,    // [BQ, 128] = (h,l,p)
                            const float* __restrict__ ref,   // [BQ, 4]
                            const float* __restrict__ vr,    // [B, NL, 2]
                            float* __restrict__ out) {       // [BQ, C]
  const int blk = blockIdx.x;
  const int d  = threadIdx.x;
  const int h  = blk % HH;
  const int bq = blk / HH;
  const int b  = bq / QQ;
  const int lvlH[NLV] = {80, 40, 20, 10};
  const int lvlW[NLV] = {80, 40, 20, 10};
  const int lvlS[NLV] = {0, 6400, 8000, 8400};
  float acc = 0.f;
#pragma unroll
  for (int l = 0; l < NLV; ++l) {
    const int Hl = lvlH[l], Wl = lvlW[l], st = lvlS[l];
    const float vrx = vr[(b * NLV + l) * 2 + 0];
    const float vry = vr[(b * NLV + l) * 2 + 1];
    const float rx = ref[bq * 4 + 0] * vrx;
    const float ry = ref[bq * 4 + 1] * vry;
    const float rw = ref[bq * 4 + 2] * vrx;
    const float rh = ref[bq * 4 + 3] * vry;
#pragma unroll
    for (int p = 0; p < PP; ++p) {
      const int oidx = ((bq * HH + h) * NLV + l) * PP + p;
      const float lx = rx + off[oidx * 2 + 0] * (1.f / PP) * rw * 0.5f;
      const float ly = ry + off[oidx * 2 + 1] * (1.f / PP) * rh * 0.5f;
      const float w  = aw[oidx];
      const float x = lx * Wl - 0.5f;
      const float y = ly * Hl - 0.5f;
      const float x0 = floorf(x), y0 = floorf(y);
      const float fx = x - x0, fy = y - y0;
      float samp = 0.f;
#pragma unroll
      for (int dy = 0; dy < 2; ++dy) {
#pragma unroll
        for (int dx = 0; dx < 2; ++dx) {
          const int xi = (int)x0 + dx;
          const int yi = (int)y0 + dy;
          if (xi >= 0 && xi < Wl && yi >= 0 && yi < Hl) {
            const float wgt = (dx ? fx : 1.f - fx) * (dy ? fy : 1.f - fy);
            samp += wgt * val[((size_t)(b * SS + st + yi * Wl + xi)) * CC + h * HD + d];
          }
        }
      }
      acc += w * samp;
    }
  }
  out[(size_t)bq * CC + h * HD + d] = acc;
}

// boxes = sigmoid(head + inverse_sigmoid(ref)); also becomes next ref
__global__ void box_fin_kernel(const float* __restrict__ x, float* __restrict__ ref,
                               float* __restrict__ boxes_out) {
  int i = blockIdx.x * blockDim.x + threadIdx.x;
  if (i >= BQ * 4) return;
  float r = fminf(fmaxf(ref[i], 0.f), 1.f);
  float inv = logf(fmaxf(r, 1e-5f)) - logf(fmaxf(1.f - r, 1e-5f));
  float p = 1.f / (1.f + expf(-(x[i] + inv)));
  boxes_out[i] = p;
  ref[i] = p;
}

// ---------------------------------------------------------------------------
// Host-side helpers
// ---------------------------------------------------------------------------
static void cvt(const float* x, _Float16* o, int n, hipStream_t s) {
  cvt_f16_kernel<<<(n + 255) / 256, 256, 0, s>>>(x, o, n);
}

// A already converted to f16. Packs W each call, then runs the WMMA GEMM.
static void gemm16(const _Float16* A16, const float* W, const float* bias, float* out,
                   int M, int N, int K, int act, _Float16* Wt, hipStream_t s) {
  const int tilesN = (N + 15) / 16;
  const int Npad = tilesN * 16;
  const int nw = Npad * K;
  pack_w_kernel<<<(nw + 255) / 256, 256, 0, s>>>(W, Wt, K, N, Npad);
  if (tilesN % 4 == 0) {
    const int nG = tilesN / 4;
    dim3 grid((nG + 3) / 4, M / 16);
    wmma_gemm_kernel<4><<<grid, dim3(128), 0, s>>>(A16, Wt, bias, out, M, N, K, act);
  } else {
    const int nG = tilesN;
    dim3 grid((nG + 3) / 4, M / 16);
    wmma_gemm_kernel<1><<<grid, dim3(128), 0, s>>>(A16, Wt, bias, out, M, N, K, act);
  }
}

static void gemm(const float* A, const float* W, const float* bias, float* out,
                 int M, int N, int K, int act, _Float16* A16, _Float16* Wt, hipStream_t s) {
  cvt(A, A16, M * K, s);
  gemm16(A16, W, bias, out, M, N, K, act, Wt, s);
}

extern "C" void kernel_launch(void* const* d_in, const int* in_sizes, int n_in,
                              void* d_out, int out_size, void* d_ws, size_t ws_size,
                              hipStream_t stream) {
  const float* target = (const float*)d_in[0];
  const float* rpu    = (const float*)d_in[1];
  const float* memory = (const float*)d_in[2];
  const float* vr     = (const float*)d_in[3];
  const float* sa_Wqkv = (const float*)d_in[4];
  const float* sa_bqkv = (const float*)d_in[5];
  const float* sa_Wo   = (const float*)d_in[6];
  const float* sa_bo   = (const float*)d_in[7];
  const float* ca_Woff = (const float*)d_in[8];
  const float* ca_boff = (const float*)d_in[9];
  const float* ca_Wattn= (const float*)d_in[10];
  const float* ca_battn= (const float*)d_in[11];
  const float* ca_Wv   = (const float*)d_in[12];
  const float* ca_bv   = (const float*)d_in[13];
  const float* ca_Wout = (const float*)d_in[14];
  const float* ca_bout = (const float*)d_in[15];
  const float* ffn_W1  = (const float*)d_in[16];
  const float* ffn_b1  = (const float*)d_in[17];
  const float* ffn_W2  = (const float*)d_in[18];
  const float* ffn_b2  = (const float*)d_in[19];
  const float* ln1_g = (const float*)d_in[20];
  const float* ln1_b = (const float*)d_in[21];
  const float* ln2_g = (const float*)d_in[22];
  const float* ln2_b = (const float*)d_in[23];
  const float* ln3_g = (const float*)d_in[24];
  const float* ln3_b = (const float*)d_in[25];
  const float* norm_g = (const float*)d_in[26];
  const float* norm_b = (const float*)d_in[27];
  const float* qpos_W1 = (const float*)d_in[28];
  const float* qpos_b1 = (const float*)d_in[29];
  const float* qpos_W2 = (const float*)d_in[30];
  const float* qpos_b2 = (const float*)d_in[31];
  const float* cls_W = (const float*)d_in[32];
  const float* cls_b = (const float*)d_in[33];
  const float* box_W1 = (const float*)d_in[34];
  const float* box_b1 = (const float*)d_in[35];
  const float* box_W2 = (const float*)d_in[36];
  const float* box_b2 = (const float*)d_in[37];
  const float* box_W3 = (const float*)d_in[38];
  const float* box_b3 = (const float*)d_in[39];

  float* out = (float*)d_out;
  float* boxes_out  = out;                       // [L, B, Q, 4]
  float* logits_out = out + (size_t)LL * BQ * 4; // [L, B, Q, NC]

  // workspace carve-up (floats; all sizes multiple of 4 -> 16B alignment kept)
  float* w = (float*)d_ws;
  size_t off_ = 0;
  auto alloc = [&](size_t n) { float* p = w + off_; off_ += n; return p; };
  float* ref    = alloc(BQ * 4);
  float* output = alloc((size_t)BQ * CC);
  float* pe     = alloc((size_t)BQ * 2 * CC);
  float* qh     = alloc((size_t)BQ * CC);
  float* qpos   = alloc((size_t)BQ * CC);
  float* qbuf   = alloc((size_t)BQ * CC);
  float* qkv    = alloc((size_t)BQ * 3 * CC);
  float* ctx    = alloc((size_t)BQ * CC);
  float* sa     = alloc((size_t)BQ * CC);
  float* offb   = alloc((size_t)BQ * CC);
  float* alog   = alloc((size_t)BQ * 128);
  float* aw     = alloc((size_t)BQ * 128);
  float* val    = alloc((size_t)MVAL * CC);
  float* ms     = alloc((size_t)BQ * CC);
  float* ca     = alloc((size_t)BQ * CC);
  float* f1     = alloc((size_t)BQ * FF);
  float* f2     = alloc((size_t)BQ * CC);
  float* no     = alloc((size_t)BQ * CC);
  float* bh1    = alloc((size_t)BQ * CC);
  float* bh2    = alloc((size_t)BQ * CC);
  float* btmp   = alloc(BQ * 4);
  // f16 scratch (sized in floats: 2 halfs per float)
  _Float16* mem16 = (_Float16*)alloc((size_t)MVAL * CC / 2);     // memory, converted once
  _Float16* A16   = (_Float16*)alloc((size_t)BQ * FF / 2);       // largest activation: BQ x 1024
  _Float16* Wt    = (_Float16*)alloc((size_t)(FF * CC) / 2);     // largest Npad*K: 1024*256
  (void)ws_size; (void)in_sizes; (void)n_in; (void)out_size;

  // ref = sigmoid(rpu); output = target; mem16 = f16(memory)
  sigmoid_kernel<<<(BQ * 4 + 255) / 256, 256, 0, stream>>>(rpu, ref, BQ * 4);
  hipMemcpyAsync(output, target, (size_t)BQ * CC * sizeof(float),
                 hipMemcpyDeviceToDevice, stream);
  cvt(memory, mem16, MVAL * CC, stream);

  const int nAdd = BQ * CC;
  for (int i = 0; i < LL; ++i) {
    // --- query positional embedding ---
    pe_kernel<<<(BQ * 2 * CC + 255) / 256, 256, 0, stream>>>(ref, vr, pe);
    gemm(pe, qpos_W1, qpos_b1, qh, BQ, CC, 2 * CC, 1, A16, Wt, stream);
    gemm(qh, qpos_W2, qpos_b2, qpos, BQ, CC, CC, 0, A16, Wt, stream);

    // --- self attention ---
    add_kernel<<<(nAdd + 255) / 256, 256, 0, stream>>>(output, qpos, qbuf, nAdd);
    gemm(qbuf, sa_Wqkv + (size_t)i * CC * 3 * CC, sa_bqkv + (size_t)i * 3 * CC,
         qkv, BQ, 3 * CC, CC, 0, A16, Wt, stream);
    mha_attn_kernel<<<BB * HH * QQ, 128, 0, stream>>>(qkv, ctx);
    gemm(ctx, sa_Wo + (size_t)i * CC * CC, sa_bo + (size_t)i * CC,
         sa, BQ, CC, CC, 0, A16, Wt, stream);
    ln_kernel<<<BQ, CC, 0, stream>>>(output, sa, ln1_g + i * CC, ln1_b + i * CC, output);

    // --- deformable cross attention ---
    add_kernel<<<(nAdd + 255) / 256, 256, 0, stream>>>(output, qpos, qbuf, nAdd);
    cvt(qbuf, A16, BQ * CC, stream);   // shared by the two projections below
    gemm16(A16, ca_Woff + (size_t)i * CC * 256, ca_boff + (size_t)i * 256,
           offb, BQ, 256, CC, 0, Wt, stream);
    gemm16(A16, ca_Wattn + (size_t)i * CC * 128, ca_battn + (size_t)i * 128,
           alog, BQ, 128, CC, 0, Wt, stream);
    softmax16_kernel<<<(BQ * HH + 255) / 256, 256, 0, stream>>>(alog, aw);
    gemm16(mem16, ca_Wv + (size_t)i * CC * CC, ca_bv + (size_t)i * CC,
           val, MVAL, CC, CC, 0, Wt, stream);
    msda_kernel<<<BQ * HH, HD, 0, stream>>>(val, offb, aw, ref, vr, ms);
    gemm(ms, ca_Wout + (size_t)i * CC * CC, ca_bout + (size_t)i * CC,
         ca, BQ, CC, CC, 0, A16, Wt, stream);
    ln_kernel<<<BQ, CC, 0, stream>>>(output, ca, ln2_g + i * CC, ln2_b + i * CC, output);

    // --- FFN ---
    gemm(output, ffn_W1 + (size_t)i * CC * FF, ffn_b1 + (size_t)i * FF,
         f1, BQ, FF, CC, 1, A16, Wt, stream);
    gemm(f1, ffn_W2 + (size_t)i * FF * CC, ffn_b2 + (size_t)i * CC,
         f2, BQ, CC, FF, 0, A16, Wt, stream);
    ln_kernel<<<BQ, CC, 0, stream>>>(output, f2, ln3_g + i * CC, ln3_b + i * CC, output);

    // --- heads ---
    ln_kernel<<<BQ, CC, 0, stream>>>(output, nullptr, norm_g, norm_b, no);
    cvt(no, A16, BQ * CC, stream);     // shared by box head stage 1 and cls head
    gemm16(A16, box_W1 + (size_t)i * CC * CC, box_b1 + (size_t)i * CC,
           bh1, BQ, CC, CC, 1, Wt, stream);
    gemm16(A16, cls_W + (size_t)i * CC * NCLS, cls_b + (size_t)i * NCLS,
           logits_out + (size_t)i * BQ * NCLS, BQ, NCLS, CC, 0, Wt, stream);
    gemm(bh1, box_W2 + (size_t)i * CC * CC, box_b2 + (size_t)i * CC,
         bh2, BQ, CC, CC, 1, A16, Wt, stream);
    gemm(bh2, box_W3 + (size_t)i * CC * 4, box_b3 + (size_t)i * 4,
         btmp, BQ, 4, CC, 0, A16, Wt, stream);
    box_fin_kernel<<<(BQ * 4 + 255) / 256, 256, 0, stream>>>(
        btmp, ref, boxes_out + (size_t)i * BQ * 4);
  }
}
